// PodCritic_29807073034551
// MI455X (gfx1250) — matrix operand order repaired and checked by reading, not compiled, not run on hardware
//
#include <hip/hip_runtime.h>
#include <hip/hip_fp16.h>

typedef __attribute__((ext_vector_type(16))) _Float16 v16h;
typedef __attribute__((ext_vector_type(8)))  _Float16 v8h;
typedef __attribute__((ext_vector_type(8)))  float    v8f;

#define B_SZ   1024
#define S_SZ   128
#define NTOT   (B_SZ * S_SZ)
#define KF     96      // padded feature dim (79 -> 96)
#define H1     256
#define H2     128
#define LH     64
#define GDIM   256     // 4*LH

// ---------------------------------------------------------------------------
// WMMA fragment loaders (CDNA5 ISA 7.12.2 layouts, wave32)
// ---------------------------------------------------------------------------
// A-matrix 16x32 f16, row-major source with row stride lda (elements).
// VGPRs 0-3 hold K = k0 + 8*half + {0..7} (contiguous), VGPRs 4-7 K += 16.
// => two 16-byte vector loads per lane.
__device__ inline v16h load_a_frag(const _Float16* base, int lda, int k0) {
  union { v16h full; v8h part[2]; } r;
  const int lane = threadIdx.x & 31;
  const int m = lane & 15, hf = lane >> 4;
  const _Float16* p = base + (size_t)m * lda + k0 + 8 * hf;
  r.part[0] = *(const v8h*)(p);
  r.part[1] = *(const v8h*)(p + 16);
  return r.full;
}

// B-matrix fragment from pre-packed buffer: fragment (kc, nt) stored as
// 32 lanes x 16 halves contiguous => one 32-byte vector load per lane.
__device__ inline v16h load_b_packed(const _Float16* pack, int ntCount,
                                     int kc, int nt) {
  const int lane = threadIdx.x & 31;
  return *(const v16h*)(pack +
      (((size_t)kc * ntCount + nt) * 32 + lane) * 16);
}

__device__ inline float wsum32(float v) {
  v += __shfl_xor(v, 16, 32); v += __shfl_xor(v, 8, 32);
  v += __shfl_xor(v, 4, 32);  v += __shfl_xor(v, 2, 32);
  v += __shfl_xor(v, 1, 32);
  return v;
}
__device__ inline float hsum16(float v) {   // reduce within 16-lane half
  v += __shfl_xor(v, 8, 16); v += __shfl_xor(v, 4, 16);
  v += __shfl_xor(v, 2, 16); v += __shfl_xor(v, 1, 16);
  return v;
}
__device__ inline float sigm(float x) { return 1.0f / (1.0f + __expf(-x)); }

// Pack W[K,N] (f32 row-major) into WMMA B-fragment order, zero-padding
// rows >= Kvalid. Element t encodes (chunk, lane, elem); chunk = kc*(N/16)+nt.
__device__ inline void pack_b_mat(const float* W, _Float16* dst, int Kvalid,
                                  int N, int total, int i, int stride) {
  const int ntCount = N >> 4;
  for (int t = i; t < total; t += stride) {
    int chunk = t >> 9;            // / 512
    int within = t & 511;
    int lane = within >> 4;
    int e = within & 15;
    int kc = chunk / ntCount, nt = chunk % ntCount;
    int k = kc * 32 + 8 * (lane >> 4) + (e & 7) + 16 * (e >> 3);
    int n = nt * 16 + (lane & 15);
    dst[t] = (k < Kvalid) ? (_Float16)W[(size_t)k * N + n] : (_Float16)0.0f;
  }
}

// ---------------------------------------------------------------------------
// Kernel 0: pack weights to WMMA-order f16, fold gate biases, init h/c.
// ---------------------------------------------------------------------------
__global__ void prep_kernel(const float* enc_w1, const float* enc_w2,
                            const float* ih_w, const float* hh_w,
                            const float* ih_b, const float* hh_b,
                            const float* h0, const float* c0,
                            _Float16* w1p, _Float16* w2p,
                            _Float16* ihp, _Float16* hhp,
                            _Float16* h16, float* cbuf, float* gbias) {
  int i = blockIdx.x * blockDim.x + threadIdx.x;
  int stride = gridDim.x * blockDim.x;
  pack_b_mat(enc_w1, w1p, 79, H1, KF * H1, i, stride);
  pack_b_mat(enc_w2, w2p, H1, H2, H1 * H2, i, stride);
  pack_b_mat(ih_w, ihp, H2, GDIM, H2 * GDIM, i, stride);
  pack_b_mat(hh_w, hhp, LH, GDIM, LH * GDIM, i, stride);
  for (int t = i; t < GDIM; t += stride) gbias[t] = ih_b[t] + hh_b[t];
  for (int t = i; t < B_SZ * LH; t += stride) {
    h16[t] = (_Float16)h0[t];
    cbuf[t] = c0[t];
  }
}

// ---------------------------------------------------------------------------
// Kernel 1: map encoder (wave-per-sample, lane = channel) + feature build.
// Writes feat [NTOT, 96] f16 (cols 79..95 zero).
// ---------------------------------------------------------------------------
__global__ __launch_bounds__(256) void map_kernel(
    const float* self_obs, const float* tm_obs, const float* en_obs,
    const float* cp_obs, const float* map_obs,
    const float* mpw, const float* mpb,
    const float* wq, const float* bq, const float* wk, const float* bk,
    const float* wv, const float* bv, const float* wo, const float* bo,
    const float* g1, const float* bb1, const float* fw1, const float* fb1,
    const float* fw2, const float* fb2, const float* g2, const float* bb2,
    _Float16* feat) {
  __shared__ float s_wq[1024], s_wk[1024], s_wv[1024], s_wo[1024];
  __shared__ float s_w1[2048], s_w2[2048];
  __shared__ float s_sm[448];
  const int tid = threadIdx.x;
  for (int idx = tid; idx < 1024; idx += 256) {
    s_wq[idx] = wq[idx]; s_wk[idx] = wk[idx];
    s_wv[idx] = wv[idx]; s_wo[idx] = wo[idx];
  }
  for (int idx = tid; idx < 2048; idx += 256) { s_w1[idx] = fw1[idx]; s_w2[idx] = fw2[idx]; }
  if (tid < 64) s_sm[tid] = mpw[tid];
  if (tid < 32) {
    s_sm[64 + tid] = mpb[tid];  s_sm[96 + tid] = bq[tid];
    s_sm[128 + tid] = bk[tid];  s_sm[160 + tid] = bv[tid];
    s_sm[192 + tid] = bo[tid];  s_sm[224 + tid] = g1[tid];
    s_sm[256 + tid] = bb1[tid]; s_sm[352 + tid] = fb2[tid];
    s_sm[384 + tid] = g2[tid];  s_sm[416 + tid] = bb2[tid];
  }
  if (tid < 64) s_sm[288 + tid] = fb1[tid];
  __syncthreads();

  const int wid = tid >> 5, c = tid & 31;
  const int i = blockIdx.x * 8 + wid;   // sample index, exact cover
  const float* mo = map_obs + (size_t)i * 12;

  float x[6], q[6], kk[6], vv[6];
#pragma unroll
  for (int t = 0; t < 6; ++t) {
    float m0 = mo[2 * t], m1 = mo[2 * t + 1];
    x[t] = fmaxf(m0 * s_sm[c] + m1 * s_sm[32 + c] + s_sm[64 + c], 0.0f);
    q[t] = s_sm[96 + c]; kk[t] = s_sm[128 + c]; vv[t] = s_sm[160 + c];
  }
  for (int k = 0; k < 32; ++k) {
    float a = s_wq[k * 32 + c], b = s_wk[k * 32 + c], d = s_wv[k * 32 + c];
#pragma unroll
    for (int t = 0; t < 6; ++t) {
      float xb = __shfl(x[t], k, 32);
      q[t] += xb * a; kk[t] += xb * b; vv[t] += xb * d;
    }
  }
  // attention: heads = lanes [0,16) and [16,32)
  float o[6];
#pragma unroll
  for (int t = 0; t < 6; ++t) {
    float s[6];
#pragma unroll
    for (int u = 0; u < 6; ++u) s[u] = hsum16(q[t] * kk[u]) * 0.25f;
    float mx = s[0];
#pragma unroll
    for (int u = 1; u < 6; ++u) mx = fmaxf(mx, s[u]);
    float den = 0.0f;
#pragma unroll
    for (int u = 0; u < 6; ++u) { s[u] = __expf(s[u] - mx); den += s[u]; }
    float inv = 1.0f / den;
    float acc = 0.0f;
#pragma unroll
    for (int u = 0; u < 6; ++u) acc += s[u] * inv * vv[u];
    o[t] = acc;
  }
  // o @ wo + bo, residual, LN1
  float y[6];
  {
    float ao[6];
#pragma unroll
    for (int t = 0; t < 6; ++t) ao[t] = s_sm[192 + c];
    for (int k = 0; k < 32; ++k) {
      float w = s_wo[k * 32 + c];
#pragma unroll
      for (int t = 0; t < 6; ++t) ao[t] += __shfl(o[t], k, 32) * w;
    }
#pragma unroll
    for (int t = 0; t < 6; ++t) {
      float r = x[t] + ao[t];
      float mu = wsum32(r) * (1.0f / 32.0f);
      float d = r - mu;
      float var = wsum32(d * d) * (1.0f / 32.0f);
      y[t] = d * rsqrtf(var + 1e-5f) * s_sm[224 + c] + s_sm[256 + c];
    }
  }
  // FFN 32->64->32, residual, LN2, max over tokens
  float memb;
  {
    float hA[6], hB[6];
#pragma unroll
    for (int t = 0; t < 6; ++t) { hA[t] = s_sm[288 + c]; hB[t] = s_sm[288 + 32 + c]; }
    for (int k = 0; k < 32; ++k) {
      float wa = s_w1[k * 64 + c], wb = s_w1[k * 64 + 32 + c];
#pragma unroll
      for (int t = 0; t < 6; ++t) {
        float yb = __shfl(y[t], k, 32);
        hA[t] += yb * wa; hB[t] += yb * wb;
      }
    }
#pragma unroll
    for (int t = 0; t < 6; ++t) { hA[t] = fmaxf(hA[t], 0.0f); hB[t] = fmaxf(hB[t], 0.0f); }
    float f[6];
#pragma unroll
    for (int t = 0; t < 6; ++t) f[t] = s_sm[352 + c];
    for (int k = 0; k < 32; ++k) {
      float wa = s_w2[k * 32 + c], wb = s_w2[(32 + k) * 32 + c];
#pragma unroll
      for (int t = 0; t < 6; ++t)
        f[t] += __shfl(hA[t], k, 32) * wa + __shfl(hB[t], k, 32) * wb;
    }
    memb = -1e30f;
#pragma unroll
    for (int t = 0; t < 6; ++t) {
      float r = y[t] + f[t];
      float mu = wsum32(r) * (1.0f / 32.0f);
      float d = r - mu;
      float var = wsum32(d * d) * (1.0f / 32.0f);
      memb = fmaxf(memb, d * rsqrtf(var + 1e-5f) * s_sm[384 + c] + s_sm[416 + c]);
    }
  }
  // build 96-wide feature row (f16)
  const float* so = self_obs + (size_t)i * 15;
  const float* to = tm_obs + (size_t)i * 13;
  const float* eo = en_obs + (size_t)i * 26;
  const float* po = cp_obs + (size_t)i * 6;
  _Float16* fr = feat + (size_t)i * KF;
#pragma unroll
  for (int rep = 0; rep < 3; ++rep) {
    int idx = c + 32 * rep;
    int src = idx - 47; src = src < 0 ? 0 : (src > 31 ? 31 : src);
    float mb = __shfl(memb, src, 32);      // uniform wave execution
    float v;
    if (idx < 15)      v = so[idx];
    else if (idx < 28) v = to[idx - 15];
    else if (idx < 41) v = 0.5f * (eo[idx - 28] + eo[13 + idx - 28]);
    else if (idx < 47) v = po[idx - 41];
    else if (idx < 79) v = mb;
    else               v = 0.0f;
    fr[idx] = (_Float16)v;
  }
}

// ---------------------------------------------------------------------------
// Kernel 2: fused MLP 96->256 (relu) ->128 (relu), WMMA, 16 rows / block.
// ---------------------------------------------------------------------------
__global__ __launch_bounds__(256) void mlp_kernel(
    const _Float16* feat, const _Float16* w1p, const _Float16* w2p,
    const float* b1, const float* b2, _Float16* emb) {
  __shared__ __align__(16) _Float16 s_act[16 * H1];
  const int row0 = blockIdx.x * 16;
  const int wid = threadIdx.x >> 5, lane = threadIdx.x & 31;
  const int n = lane & 15, hf = lane >> 4;
  const _Float16* abase = feat + (size_t)row0 * KF;

#pragma unroll
  for (int sub = 0; sub < 2; ++sub) {
    int n0 = wid * 32 + sub * 16;
    v8f acc = {};
#pragma unroll
    for (int kc = 0; kc < 3; ++kc) {
      v16h a = load_a_frag(abase, KF, kc * 32);
      v16h b = load_b_packed(w1p, H1 / 16, kc, n0 >> 4);
      acc = __builtin_amdgcn_wmma_f32_16x16x32_f16(false, a, false, b,
                                                   (short)0, acc, false, false);
    }
    float bias = b1[n0 + n];
#pragma unroll
    for (int r = 0; r < 8; ++r) {
      int m = r + 8 * hf;
      s_act[m * H1 + n0 + n] = (_Float16)fmaxf(acc[r] + bias, 0.0f);
    }
  }
  __syncthreads();
  {
    int n0 = wid * 16;
    v8f acc = {};
#pragma unroll
    for (int kc = 0; kc < 8; ++kc) {
      v16h a = load_a_frag(s_act, H1, kc * 32);
      v16h b = load_b_packed(w2p, H2 / 16, kc, wid);
      acc = __builtin_amdgcn_wmma_f32_16x16x32_f16(false, a, false, b,
                                                   (short)0, acc, false, false);
    }
    float bias = b2[n0 + n];
#pragma unroll
    for (int r = 0; r < 8; ++r) {
      int m = r + 8 * hf;
      emb[(size_t)(row0 + m) * H2 + n0 + n] = (_Float16)fmaxf(acc[r] + bias, 0.0f);
    }
  }
}

// ---------------------------------------------------------------------------
// Kernel 3: one LSTM step. gates = emb_t@ih + h@hh (+bias), pointwise,
// value head. 16 batch rows / block, 8 waves x 2 col tiles.
// ---------------------------------------------------------------------------
__global__ __launch_bounds__(256) void lstm_step_kernel(
    const _Float16* emb, const _Float16* ihp, const _Float16* hhp,
    const float* gbias, const float* val_w, const float* val_b,
    _Float16* h16, float* cbuf, float* out, int t) {
  __shared__ float s_g[16 * GDIM];
  __shared__ float s_h[16 * LH];
  const int b0 = blockIdx.x * 16;
  const int wid = threadIdx.x >> 5, lane = threadIdx.x & 31;
  const int n = lane & 15, hf = lane >> 4;
  const _Float16* aemb = emb + ((size_t)b0 * S_SZ + t) * H2;  // row stride S*H2
  const _Float16* ah = h16 + (size_t)b0 * LH;

#pragma unroll
  for (int sub = 0; sub < 2; ++sub) {
    int n0 = wid * 32 + sub * 16;
    int nt = n0 >> 4;
    v8f acc = {};
#pragma unroll
    for (int kc = 0; kc < 4; ++kc) {
      v16h a = load_a_frag(aemb, S_SZ * H2, kc * 32);
      v16h b = load_b_packed(ihp, GDIM / 16, kc, nt);
      acc = __builtin_amdgcn_wmma_f32_16x16x32_f16(false, a, false, b,
                                                   (short)0, acc, false, false);
    }
#pragma unroll
    for (int kc = 0; kc < 2; ++kc) {
      v16h a = load_a_frag(ah, LH, kc * 32);
      v16h b = load_b_packed(hhp, GDIM / 16, kc, nt);
      acc = __builtin_amdgcn_wmma_f32_16x16x32_f16(false, a, false, b,
                                                   (short)0, acc, false, false);
    }
#pragma unroll
    for (int r = 0; r < 8; ++r) s_g[(r + 8 * hf) * GDIM + n0 + n] = acc[r];
  }
  if (t + 1 < S_SZ) __builtin_prefetch(aemb + H2, 0, 0);  // next step's A tile
  __syncthreads();

  const int row = threadIdx.x >> 4;
  const int j0 = (threadIdx.x & 15) * 4;
#pragma unroll
  for (int p = 0; p < 4; ++p) {
    int j = j0 + p;
    int gb = row * GDIM;
    float gi = s_g[gb + j]       + gbias[j];
    float gf = s_g[gb + 64 + j]  + gbias[64 + j];
    float gg = s_g[gb + 128 + j] + gbias[128 + j];
    float go = s_g[gb + 192 + j] + gbias[192 + j];
    size_t ci = (size_t)(b0 + row) * LH + j;
    float cold = cbuf[ci];
    float cn = sigm(gf) * cold + sigm(gi) * tanhf(gg);
    float hn = sigm(go) * tanhf(cn);
    cbuf[ci] = cn;
    h16[ci] = (_Float16)hn;
    s_h[row * LH + j] = hn;
    if (t == S_SZ - 1) {
      out[NTOT + ci] = hn;                      // hn output
      out[NTOT + (size_t)B_SZ * LH + ci] = cn;  // cn output
    }
  }
  __syncthreads();
  if (threadIdx.x < 16) {
    float acc = val_b[0];
    for (int j = 0; j < LH; ++j) acc += s_h[threadIdx.x * LH + j] * val_w[j];
    out[(size_t)(b0 + threadIdx.x) * S_SZ + t] = acc;
  }
}

// ---------------------------------------------------------------------------
extern "C" void kernel_launch(void* const* d_in, const int* in_sizes, int n_in,
                              void* d_out, int out_size, void* d_ws, size_t ws_size,
                              hipStream_t stream) {
  const float* self_obs = (const float*)d_in[0];
  const float* tm_obs   = (const float*)d_in[1];
  const float* en_obs   = (const float*)d_in[2];
  const float* cp_obs   = (const float*)d_in[3];
  const float* map_obs  = (const float*)d_in[4];
  const float* h0       = (const float*)d_in[5];
  const float* c0       = (const float*)d_in[6];
  const float* mpw = (const float*)d_in[7],  *mpb = (const float*)d_in[8];
  const float* wq  = (const float*)d_in[9],  *bq  = (const float*)d_in[10];
  const float* wk  = (const float*)d_in[11], *bk  = (const float*)d_in[12];
  const float* wv  = (const float*)d_in[13], *bv  = (const float*)d_in[14];
  const float* wo  = (const float*)d_in[15], *bo  = (const float*)d_in[16];
  const float* g1  = (const float*)d_in[17], *bb1 = (const float*)d_in[18];
  const float* fw1 = (const float*)d_in[19], *fb1 = (const float*)d_in[20];
  const float* fw2 = (const float*)d_in[21], *fb2 = (const float*)d_in[22];
  const float* g2  = (const float*)d_in[23], *bb2 = (const float*)d_in[24];
  const float* enc_w1 = (const float*)d_in[25], *enc_b1 = (const float*)d_in[26];
  const float* enc_w2 = (const float*)d_in[27], *enc_b2 = (const float*)d_in[28];
  const float* ih_w = (const float*)d_in[29], *ih_b = (const float*)d_in[30];
  const float* hh_w = (const float*)d_in[31], *hh_b = (const float*)d_in[32];
  const float* val_w = (const float*)d_in[33], *val_b = (const float*)d_in[34];
  float* out = (float*)d_out;

  char* ws = (char*)d_ws;
  size_t off = 0;
  auto alloc = [&](size_t bytes) -> void* {
    void* p = ws + off;
    off = (off + bytes + 255) & ~(size_t)255;
    return p;
  };
  _Float16* feat = (_Float16*)alloc((size_t)NTOT * KF * 2);
  _Float16* emb  = (_Float16*)alloc((size_t)NTOT * H2 * 2);
  _Float16* w1p  = (_Float16*)alloc((size_t)KF * H1 * 2);
  _Float16* w2p  = (_Float16*)alloc((size_t)H1 * H2 * 2);
  _Float16* ihp  = (_Float16*)alloc((size_t)H2 * GDIM * 2);
  _Float16* hhp  = (_Float16*)alloc((size_t)LH * GDIM * 2);
  _Float16* h16  = (_Float16*)alloc((size_t)B_SZ * LH * 2);
  float*    cbuf = (float*)alloc((size_t)B_SZ * LH * 4);
  float*    gbias = (float*)alloc((size_t)GDIM * 4);

  prep_kernel<<<256, 256, 0, stream>>>(enc_w1, enc_w2, ih_w, hh_w, ih_b, hh_b,
                                       h0, c0, w1p, w2p, ihp, hhp, h16, cbuf,
                                       gbias);
  map_kernel<<<NTOT / 8, 256, 0, stream>>>(
      self_obs, tm_obs, en_obs, cp_obs, map_obs, mpw, mpb,
      wq, bq, wk, bk, wv, bv, wo, bo, g1, bb1, fw1, fb1, fw2, fb2, g2, bb2, feat);
  mlp_kernel<<<NTOT / 16, 256, 0, stream>>>(feat, w1p, w2p, enc_b1, enc_b2, emb);
  for (int t = 0; t < S_SZ; ++t) {
    lstm_step_kernel<<<B_SZ / 16, 256, 0, stream>>>(
        emb, ihp, hhp, gbias, val_w, val_b, h16, cbuf, out, t);
  }
}
